// Decoder_32744830665090
// MI455X (gfx1250) — matrix operand order repaired and checked by reading, not compiled, not run on hardware
//
#include <hip/hip_runtime.h>

// ---------------- static dims (match reference) ----------------
#define Bc      32
#define NX_PER  1024
#define NY_PER  512
#define NXc     (Bc * NX_PER)     // 32768
#define NYc     (Bc * NY_PER)     // 16384
#define Ec      65536
#define Fc      256
#define EMBc    256
#define Hc      512
#define VOCABc  512
#define NRELc   7

typedef __attribute__((ext_vector_type(16))) __bf16 v16bf;
typedef __attribute__((ext_vector_type(8)))  float  v8f;
typedef __attribute__((ext_vector_type(8)))  __bf16 bf16x8;

// ---------------- CDNA5 async-to-LDS helpers ----------------
// LDS offset = low 32 bits of the generic pointer (LDS aperture maps addr[31:0]).
__device__ __forceinline__ void async_load_b128(unsigned lds_off, const void* gaddr) {
    asm volatile("global_load_async_to_lds_b128 %0, %1, off"
                 :: "v"(lds_off), "v"(gaddr) : "memory");
}
__device__ __forceinline__ void wait_async0() {
    asm volatile("s_wait_asynccnt 0x0" ::: "memory");
}

// ---------------- generic bf16 WMMA GEMM ----------------
// C[M,N] = act( A@B + bias + addend ). A,B bf16 row-major; fp32 accumulate.
// Optional fp32 output C and/or bf16 output C2.
// BM=128, BN=64, BK=32; 128 threads = 4 wave32; wave w computes rows [w*32, w*32+32).
// MODE: 0 = none, 1 = +bias, 2 = +bias +addend +relu
struct GemmP {
    const __bf16* A; long sA;
    const __bf16* B; long sB;
    float*        C;  long sC;    // may be null
    __bf16*       C2; long sC2;   // may be null
    const float*  bias;
    const float*  addend;         // fp32, layout [M, ldc]
    int M, N, K;
    int lda, ldb, ldc, ldc2;
};

#define LDSK 40   // padded bf16 row stride (80B rows; 16B-aligned chunks)

template <int MODE>
__global__ __launch_bounds__(128)
void k_gemm(GemmP p) {
    __shared__ __attribute__((aligned(16))) __bf16 As[128 * LDSK];  // [m][k]
    __shared__ __attribute__((aligned(16))) __bf16 Bs[64 * LDSK];   // transposed [n][k]

    const __bf16* A  = p.A + (long)blockIdx.z * p.sA;
    const __bf16* Bm = p.B + (long)blockIdx.z * p.sB;

    const int m0   = blockIdx.y * 128;
    const int n0   = blockIdx.x * 64;
    const int tid  = threadIdx.x;
    const int wave = tid >> 5;
    const int lane = tid & 31;
    const int ml   = lane & 15;
    const int half = lane >> 4;

    v8f acc[2][4] = {};

    for (int k0 = 0; k0 < p.K; k0 += 32) {
        // --- A tile 128x32 bf16 via async DMA to LDS (4 x b128 per thread) ---
        #pragma unroll
        for (int i = 0; i < 4; ++i) {
            int idx = tid + i * 128;            // 512 chunks of 8 bf16
            int r   = idx >> 2;                 // 4 chunks per 32-wide row
            int kc  = (idx & 3) * 8;
            const __bf16* g = A + (long)(m0 + r) * p.lda + k0 + kc;
            unsigned loff = (unsigned)(size_t)&As[r * LDSK + kc];
            async_load_b128(loff, g);
            __builtin_prefetch(g + 32, 0, 0);   // next K tile -> global_prefetch_b8
        }
        // --- B tile 32x64 bf16, transposed into Bs[n][k] ---
        #pragma unroll
        for (int i = 0; i < 2; ++i) {
            int idx = tid + i * 128;            // 256 chunks of 8 bf16
            int kr  = idx >> 3;                 // 8 chunks per 64-wide row
            int nc  = (idx & 7) * 8;
            const __bf16* g = Bm + (long)(k0 + kr) * p.ldb + n0 + nc;
            bf16x8 v = *(const bf16x8*)g;
            __builtin_prefetch(g + (long)32 * p.ldb, 0, 0);
            #pragma unroll
            for (int j = 0; j < 8; ++j) Bs[(nc + j) * LDSK + kr] = v[j];
        }
        wait_async0();
        __syncthreads();

        // A fragments 16x32 (lanes 0-15: K=0..15; lanes 16-31: K=16..31)
        v16bf af[2];
        #pragma unroll
        for (int mt = 0; mt < 2; ++mt) {
            const __bf16* ap = &As[(wave * 32 + mt * 16 + ml) * LDSK];
            #pragma unroll
            for (int v = 0; v < 8; ++v) {
                int kb = half * 16 + v * 2;
                af[mt][2 * v]     = ap[kb];
                af[mt][2 * v + 1] = ap[kb + 1];
            }
        }
        // 4 B fragments 32x16 (lanes 0-15 K=0..15; lanes 16-31 K=16..31); each feeds 2 WMMAs
        #pragma unroll
        for (int nt = 0; nt < 4; ++nt) {
            v16bf bfv;
            const __bf16* bp = &Bs[(nt * 16 + ml) * LDSK + half * 16];
            #pragma unroll
            for (int v = 0; v < 16; ++v) bfv[v] = bp[v];
            acc[0][nt] = __builtin_amdgcn_wmma_f32_16x16x32_bf16(
                false, af[0], false, bfv, (short)0, acc[0][nt], false, false);
            acc[1][nt] = __builtin_amdgcn_wmma_f32_16x16x32_bf16(
                false, af[1], false, bfv, (short)0, acc[1][nt], false, false);
        }
        __syncthreads();
    }

    // epilogue: C/D layout -> VGPR r holds M = r + half*8, N = lane%16
    float*  C  = p.C  ? p.C  + (long)blockIdx.z * p.sC  : nullptr;
    __bf16* C2 = p.C2 ? p.C2 + (long)blockIdx.z * p.sC2 : nullptr;
    #pragma unroll
    for (int mt = 0; mt < 2; ++mt) {
        const int mrow0 = m0 + wave * 32 + mt * 16 + half * 8;
        #pragma unroll
        for (int nt = 0; nt < 4; ++nt) {
            int n = n0 + nt * 16 + ml;
            float bv = (MODE >= 1) ? p.bias[n] : 0.0f;
            #pragma unroll
            for (int r = 0; r < 8; ++r) {
                long m = mrow0 + r;
                float v = acc[mt][nt][r] + bv;
                if (MODE == 2) v += p.addend[m * p.ldc + n];
                if (MODE == 2) v = fmaxf(v, 0.0f);
                if (C)  C[m * p.ldc + n]   = v;
                if (C2) C2[m * p.ldc2 + n] = (__bf16)v;
            }
        }
    }
}

// ---------------- small helper kernels ----------------
__global__ void k_cast_bf16(const float* __restrict__ in, __bf16* __restrict__ out, long n) {
    long i = (long)blockIdx.x * 256 + threadIdx.x;
    if (i < n) out[i] = (__bf16)in[i];
}

__global__ void k_embed(const int* __restrict__ tgt_y, const float* __restrict__ tab,
                        __bf16* __restrict__ y0b) {
    long i = (long)blockIdx.x * 256 + threadIdx.x;     // over NY*256
    int row = (int)(i >> 8), col = (int)(i & 255);
    y0b[i] = (__bf16)tab[(long)tgt_y[row] * EMBc + col];
}

__global__ void k_transpose_x(const float* __restrict__ x, __bf16* __restrict__ xbT) {
    long i = (long)blockIdx.x * 256 + threadIdx.x;     // over NX*F
    long b = i >> 18;                                  // 256*1024
    long rem = i & 262143;
    long f = rem >> 10, n = rem & 1023;
    xbT[i] = (__bf16)x[((b << 10) + n) * Fc + f];
}

__global__ void k_scatter(const int* __restrict__ ei, const int* __restrict__ et,
                          const float* __restrict__ proj, float* __restrict__ agg,
                          int kout, int t) {
    int e = blockIdx.x;
    if (et[e] != t) return;
    int j   = blockIdx.y * 256 + threadIdx.x;
    int src = ei[e];
    int dst = ei[Ec + e];
    atomicAdd(&agg[(long)dst * kout + j], proj[(long)src * kout + j]);
}

// scaled softmax over rows of 1024, in place (fp32) + bf16 copy for next GEMM
__global__ void k_softmax(float* __restrict__ a, __bf16* __restrict__ ab) {
    __shared__ float red[256];
    long row = blockIdx.x;
    float*  p  = a  + row * 1024;
    __bf16* p2 = ab + row * 1024;
    int t = threadIdx.x;
    float v[4];
    float mx = -1e30f;
    #pragma unroll
    for (int i = 0; i < 4; ++i) { v[i] = p[t + i * 256] * 0.0625f; mx = fmaxf(mx, v[i]); }
    red[t] = mx; __syncthreads();
    for (int s = 128; s > 0; s >>= 1) { if (t < s) red[t] = fmaxf(red[t], red[t + s]); __syncthreads(); }
    mx = red[0]; __syncthreads();
    float sum = 0.0f;
    #pragma unroll
    for (int i = 0; i < 4; ++i) { v[i] = __expf(v[i] - mx); sum += v[i]; }
    red[t] = sum; __syncthreads();
    for (int s = 128; s > 0; s >>= 1) { if (t < s) red[t] += red[t + s]; __syncthreads(); }
    float inv = 1.0f / red[0];
    #pragma unroll
    for (int i = 0; i < 4; ++i) {
        float o = v[i] * inv;
        p[t + i * 256]  = o;
        p2[t + i * 256] = (__bf16)o;
    }
}

__global__ void k_edge_rel(const int* __restrict__ ei, const float* __restrict__ y3,
                           const float* __restrict__ Wg, const float* __restrict__ bg,
                           float* __restrict__ out) {
    int wid  = threadIdx.x >> 5, lane = threadIdx.x & 31;
    long e   = (long)blockIdx.x * 8 + wid;
    int src  = ei[e];
    int dst  = ei[Ec + e];
    float part[NRELc] = {};
    for (int k = lane; k < 2 * EMBc; k += 32) {
        float v = (k < EMBc) ? y3[(long)src * EMBc + k] : y3[(long)dst * EMBc + (k - EMBc)];
        #pragma unroll
        for (int r = 0; r < NRELc; ++r) part[r] += v * Wg[k * NRELc + r];
    }
    #pragma unroll
    for (int r = 0; r < NRELc; ++r)
        for (int s = 16; s > 0; s >>= 1) part[r] += __shfl_xor(part[r], s, 32);
    if (lane == 0) {
        #pragma unroll
        for (int r = 0; r < NRELc; ++r) out[e * NRELc + r] = part[r] + bg[r];
    }
}

__global__ void k_int2float(const int* __restrict__ in, float* __restrict__ out, int n) {
    int i = blockIdx.x * 256 + threadIdx.x;
    if (i < n) out[i] = (float)in[i];
}

// ---------------- host-side launchers ----------------
static void launch_gemm(hipStream_t s, int mode,
                        const __bf16* A, long sA, int lda,
                        const __bf16* Bm, long sB, int ldb,
                        float* C, long sC, __bf16* C2, long sC2,
                        int ldc, int ldc2,
                        const float* bias, const float* addend,
                        int M, int N, int K, int batch) {
    GemmP p{A, sA, Bm, sB, C, sC, C2, sC2, bias, addend, M, N, K, lda, ldb, ldc, ldc2};
    dim3 g(N / 64, M / 128, batch);
    if (mode == 0)      k_gemm<0><<<g, 128, 0, s>>>(p);
    else if (mode == 1) k_gemm<1><<<g, 128, 0, s>>>(p);
    else                k_gemm<2><<<g, 128, 0, s>>>(p);
}

static void cast_bf16(hipStream_t s, const float* in, __bf16* out, long n) {
    k_cast_bf16<<<(unsigned)((n + 255) / 256), 256, 0, s>>>(in, out, n);
}

extern "C" void kernel_launch(void* const* d_in, const int* in_sizes, int n_in,
                              void* d_out, int out_size, void* d_ws, size_t ws_size,
                              hipStream_t stream) {
    (void)in_sizes; (void)n_in; (void)out_size; (void)ws_size;
    const float* x     = (const float*)d_in[0];
    const int*   tgt_y = (const int*)d_in[1];
    const int*   ei    = (const int*)d_in[2];
    const int*   et    = (const int*)d_in[3];
    const float* tab   = (const float*)d_in[6];
    const float* Wz    = (const float*)d_in[7];
    const float* bz    = (const float*)d_in[8];
    const float* Wg    = (const float*)d_in[9];
    const float* bg    = (const float*)d_in[10];
    const float* We[3] = {(const float*)d_in[11], (const float*)d_in[15], (const float*)d_in[19]};
    const float* Wq[3] = {(const float*)d_in[12], (const float*)d_in[16], (const float*)d_in[20]};
    const float* Wc[3] = {(const float*)d_in[13], (const float*)d_in[17], (const float*)d_in[21]};
    const float* bb[3] = {(const float*)d_in[14], (const float*)d_in[18], (const float*)d_in[22]};
    const int    kin_[3]  = {EMBc, Hc, Hc};
    const int    kout_[3] = {Hc, Hc, EMBc};

    // ---- workspace carve ----
    char* w = (char*)d_ws;
    auto alloc = [&](size_t bytes) { char* r = w; w += (bytes + 255) & ~(size_t)255; return r; };
    __bf16* wWe[3]; __bf16* wWq[3]; __bf16* wWc[3];
    for (int l = 0; l < 3; ++l) {
        wWe[l] = (__bf16*)alloc((size_t)4 * kin_[l] * kout_[l] * 2);
        wWq[l] = (__bf16*)alloc((size_t)kin_[l] * Fc * 2);
        wWc[l] = (__bf16*)alloc((size_t)Fc * kout_[l] * 2);
    }
    __bf16* wWz    = (__bf16*)alloc((size_t)EMBc * VOCABc * 2);
    __bf16* xbf    = (__bf16*)alloc((size_t)NXc * Fc * 2);
    __bf16* xbT    = (__bf16*)alloc((size_t)Bc * Fc * NX_PER * 2);
    __bf16* y0b    = (__bf16*)alloc((size_t)NYc * EMBc * 2);
    __bf16* yAb    = (__bf16*)alloc((size_t)NYc * Hc * 2);
    __bf16* yBb    = (__bf16*)alloc((size_t)NYc * Hc * 2);
    __bf16* y3b    = (__bf16*)alloc((size_t)NYc * EMBc * 2);
    __bf16* qb     = (__bf16*)alloc((size_t)NYc * Fc * 2);
    __bf16* ctxb   = (__bf16*)alloc((size_t)NYc * Fc * 2);
    __bf16* alphab = (__bf16*)alloc((size_t)Bc * NY_PER * NX_PER * 2);
    float*  proj   = (float*)alloc((size_t)NYc * Hc * 4);
    float*  agg    = (float*)alloc((size_t)NYc * Hc * 4);

    // ---- output layout (flat, reference return order) ----
    float* out   = (float*)d_out;
    float* o_y3  = out;                                      // 16384*256
    float* o_ei  = o_y3  + (long)NYc * EMBc;                 // 2*E
    float* o_et  = o_ei  + 2L * Ec;                          // E
    float* o_ys  = o_et  + (long)Ec;                         // 16384*512
    float* o_rel = o_ys  + (long)NYc * VOCABc;               // E*7
    float* o_emb = o_rel + (long)Ec * NRELc;                 // 16384*512
    float* o_a1  = o_emb + (long)NYc * VOCABc;               // 32*512*1024
    float* o_a2  = o_a1  + (long)Bc * NY_PER * NX_PER;
    float* o_a3  = o_a2  + (long)Bc * NY_PER * NX_PER;
    float* o_a[3] = {o_a1, o_a2, o_a3};
    __bf16* ybuf[3] = {yAb, yBb, y3b};
    float*  yout_f32[3] = {nullptr, nullptr, o_y3};          // only y3 needed in fp32

    // ---- one-time bf16 conversions (weights + x) ----
    for (int l = 0; l < 3; ++l) {
        cast_bf16(stream, We[l], wWe[l], (long)4 * kin_[l] * kout_[l]);
        cast_bf16(stream, Wq[l], wWq[l], (long)kin_[l] * Fc);
        cast_bf16(stream, Wc[l], wWc[l], (long)Fc * kout_[l]);
    }
    cast_bf16(stream, Wz, wWz, (long)EMBc * VOCABc);
    cast_bf16(stream, x, xbf, (long)NXc * Fc);
    k_transpose_x<<<(NXc * Fc) / 256, 256, 0, stream>>>(x, xbT);
    k_embed<<<(NYc * EMBc) / 256, 256, 0, stream>>>(tgt_y, tab, y0b);
    k_int2float<<<(2 * Ec) / 256, 256, 0, stream>>>(ei, o_ei, 2 * Ec);
    k_int2float<<<Ec / 256, 256, 0, stream>>>(et, o_et, Ec);

    // embeds_out = y0 @ Wz + bz
    launch_gemm(stream, 1, y0b, 0, EMBc, wWz, 0, VOCABc,
                o_emb, 0, nullptr, 0, VOCABc, 0, bz, nullptr, NYc, VOCABc, EMBc, 1);

    // ---- three GCN attention decoder layers ----
    const __bf16* yin = y0b;
    for (int l = 0; l < 3; ++l) {
        int kin = kin_[l], kout = kout_[l];
        // agg = segment_sum of per-edge-type projections
        hipMemsetAsync(agg, 0, (size_t)NYc * kout * sizeof(float), stream);
        for (int t = 0; t < 4; ++t) {
            launch_gemm(stream, 0, yin, 0, kin, wWe[l] + (long)t * kin * kout, 0, kout,
                        proj, 0, nullptr, 0, kout, 0, nullptr, nullptr, NYc, kout, kin, 1);
            dim3 sg(Ec, kout / 256);
            k_scatter<<<sg, 256, 0, stream>>>(ei, et, proj, agg, kout, t);
        }
        // q = yin @ Wq  (bf16 out)
        launch_gemm(stream, 0, yin, 0, kin, wWq[l], 0, Fc,
                    nullptr, 0, qb, 0, Fc, Fc, nullptr, nullptr, NYc, Fc, kin, 1);
        // raw scores -> alpha slot in d_out (fp32), batched per graph: q_b @ x_b^T
        launch_gemm(stream, 0, qb, (long)NY_PER * Fc, Fc, xbT, (long)Fc * NX_PER, NX_PER,
                    o_a[l], (long)NY_PER * NX_PER, nullptr, 0, NX_PER, 0,
                    nullptr, nullptr, NY_PER, NX_PER, Fc, Bc);
        // scaled softmax in place + bf16 copy
        k_softmax<<<Bc * NY_PER, 256, 0, stream>>>(o_a[l], alphab);
        // ctx = alpha @ x_b  (bf16 out)
        launch_gemm(stream, 0, alphab, (long)NY_PER * NX_PER, NX_PER, xbf, (long)NX_PER * Fc, Fc,
                    nullptr, 0, ctxb, (long)NY_PER * Fc, 0, Fc,
                    nullptr, nullptr, NY_PER, Fc, NX_PER, Bc);
        // yout = relu(agg + ctx @ Wc + b); bf16 always, fp32 only for y3
        launch_gemm(stream, 2, ctxb, 0, Fc, wWc[l], 0, kout,
                    yout_f32[l], 0, ybuf[l], 0, kout, kout, bb[l], agg, NYc, kout, Fc, 1);
        yin = ybuf[l];
    }

    // ---- heads ----
    // y_score = y3 @ Wz + bz
    launch_gemm(stream, 1, y3b, 0, EMBc, wWz, 0, VOCABc,
                o_ys, 0, nullptr, 0, VOCABc, 0, bz, nullptr, NYc, VOCABc, EMBc, 1);
    // y_edge_rel_score = concat(y3[src], y3[dst]) @ Wg + bg
    k_edge_rel<<<Ec / 8, 256, 0, stream>>>(ei, o_y3, Wg, bg, o_rel);
}
